// CustomMultiheadAttention_26654567039014
// MI455X (gfx1250) — matrix operand order repaired and checked by reading, not compiled, not run on hardware
//
#include <hip/hip_runtime.h>
#include <hip/hip_bf16.h>

// Shapes (compile-time constants matching the reference)
#define B_    2
#define T_    2048
#define D_    1024
#define H_    16
#define HD_   64
#define HALF_ 32
#define BT_   (B_ * T_)

// Padded LDS row stride for staged B tiles: 32 data halfs + 8 pad halfs = 80B.
// Bank for row n's first dword = (n*20) % 64 -> all 16 rows distinct.
#define BROW_ 40

typedef __attribute__((ext_vector_type(16))) _Float16 v16h;
typedef __attribute__((ext_vector_type(8)))  _Float16 v8h;
typedef __attribute__((ext_vector_type(8)))  float    v8f;
typedef _Float16 half_t;

// ---------------------------------------------------------------------------
// WMMA fragment helpers for v_wmma_f32_16x16x32_f16 (wave32).
// Layouts per CDNA5 ISA 7.12.2:
//  A (16x32): lane m<16 -> row m, K {0..7,16..23}; lane m+16 -> K {8..15,24..31}
//  B (32x16): lane n<16 -> col n, K {0..7,16..23}; lane n+16 -> K {8..15,24..31}
//  C/D:       elem r -> row r + (lane>=16 ? 8:0), col lane&15
// ---------------------------------------------------------------------------

__device__ __forceinline__ v16h load_frag_a(const half_t* __restrict__ base,
                                            int ld, int lane) {
  const int m    = lane & 15;
  const int koff = (lane & 16) ? 8 : 0;
  const half_t* p = base + m * ld + koff;
  v8h lo = *(const v8h*)(p);
  v8h hi = *(const v8h*)(p + 16);
  v16h a;
#pragma unroll
  for (int i = 0; i < 8; ++i) { a[i] = lo[i]; a[i + 8] = hi[i]; }
  return a;
}

// A-fragment sourced from an f32 matrix (convert in-register).
__device__ __forceinline__ v16h load_frag_a_f32(const float* __restrict__ base,
                                                int ld, int lane) {
  const int m    = lane & 15;
  const int koff = (lane & 16) ? 8 : 0;
  const float* p = base + m * ld + koff;
  v16h a;
#pragma unroll
  for (int i = 0; i < 8; ++i) {
    a[i]     = (half_t)p[i];
    a[i + 8] = (half_t)p[i + 16];
  }
  return a;
}

// B-fragment from global: "base" = row-major N-side matrix, row index == n.
__device__ __forceinline__ v16h load_frag_b(const half_t* __restrict__ base,
                                            int ld, int lane) {
  const int n    = lane & 15;
  const int koff = (lane & 16) ? 8 : 0;
  const half_t* p = base + n * ld + koff;
  v8h lo = *(const v8h*)(p);
  v8h hi = *(const v8h*)(p + 16);
  v16h b;
#pragma unroll
  for (int i = 0; i < 8; ++i) { b[i] = lo[i]; b[i + 8] = hi[i]; }
  return b;
}

// B-fragment from an LDS-staged [64][BROW_] tile (ds_load_b128 path).
__device__ __forceinline__ v16h load_frag_b_lds(const half_t (*sb)[BROW_],
                                                int j, int lane) {
  const int n    = lane & 15;
  const int koff = (lane & 16) ? 8 : 0;
  const half_t* p = &sb[j * 16 + n][koff];
  v8h lo = *(const v8h*)(p);
  v8h hi = *(const v8h*)(p + 16);
  v16h b;
#pragma unroll
  for (int i = 0; i < 8; ++i) { b[i] = lo[i]; b[i + 8] = hi[i]; }
  return b;
}

__device__ __forceinline__ v8f wmma16(v16h a, v16h b, v8f c) {
  return __builtin_amdgcn_wmma_f32_16x16x32_f16(
      /*neg_a=*/false, a, /*neg_b=*/false, b,
      /*c_mod=*/(short)0, c, /*reuse_a=*/false, /*reuse_b=*/false);
}

// ---------------------------------------------------------------------------
// Async staging of one 64-row x 32-half B tile into LDS (gfx1250 async path).
// 128 threads, 2 x global_load_async_to_lds_b128 per thread (ASYNCcnt += 2
// per wave per stage). Flat LDS addresses: low 32 bits of a generic pointer
// to __shared__ are the wave-relative LDS byte offset (ISA 10.2 aperture).
// ---------------------------------------------------------------------------
__device__ __forceinline__ void stage_b_tile_async(half_t (*sb)[BROW_],
                                                   const half_t* __restrict__ gbase,
                                                   int ldg, int tid) {
#pragma unroll
  for (int u = 0; u < 2; ++u) {
    const int c   = tid * 2 + u;       // 0..255 sixteen-byte chunks
    const int row = c >> 2;            // 0..63
    const int ci  = (c & 3) * 8;       // half offset within row: 0,8,16,24
    const half_t* g = gbase + (size_t)row * ldg + ci;
    const uint32_t l = (uint32_t)(uintptr_t)(&sb[row][ci]);
    asm volatile("global_load_async_to_lds_b128 %0, %1, off"
                 :
                 : "v"(l), "v"(g)
                 : "memory");
  }
}

#define WAIT_ASYNCCNT_2() asm volatile("s_wait_asynccnt 2" ::: "memory")
#define WAIT_ASYNCCNT_0() asm volatile("s_wait_asynccnt 0" ::: "memory")

// ---------------------------------------------------------------------------
// Kernel 1: f32 -> f16 conversion (x and the four weight matrices).
// ---------------------------------------------------------------------------
__global__ void cvt_f16_kernel(const float* __restrict__ src,
                               half_t* __restrict__ dst, int n) {
  int i = blockIdx.x * blockDim.x + threadIdx.x;
  const int stride = gridDim.x * blockDim.x;
  for (; i < n; i += stride) dst[i] = (half_t)src[i];
}

// ---------------------------------------------------------------------------
// Kernel 2: projection  y[t, h*64+d] = sum_k x[t,k] * W[h*64+d, k] + bias
// Block = 4 waves sharing head h (same W slice -> async LDS staging, double
// buffered). Each wave owns one 16-row t-tile. ROPE=1: RoPE epilogue, store
// [b,h,t,d] f16. ROPE=0 (V): store transposed [b,h,d,t] f16.
// ---------------------------------------------------------------------------
template <int ROPE>
__global__ void proj_kernel(const half_t* __restrict__ xh,
                            const half_t* __restrict__ Wh,
                            const float* __restrict__ bias,
                            const float* __restrict__ cosp,
                            const float* __restrict__ sinp,
                            half_t* __restrict__ outp) {
  __shared__ half_t sB[2][64][BROW_];

  const int tid   = threadIdx.x;
  const int lane  = tid & 31;
  const int wave  = tid >> 5;
  const int ttile = blockIdx.x * 4 + wave;     // 0 .. BT/16-1
  const int h     = blockIdx.y;
  const int tbase = ttile * 16;                // global row over B*T
  const int b     = tbase / T_;
  const int tloc0 = tbase - b * T_;            // row within T (16-aligned)

  const half_t* abase = xh + (size_t)tbase * D_;
  const half_t* bbase = Wh + (size_t)(h * HD_) * D_;

  v8f acc[4] = {};
  stage_b_tile_async(sB[0], bbase, D_, tid);
  for (int kk = 0; kk < D_ / 32; ++kk) {
    const int cur = kk & 1;
    if (kk + 1 < D_ / 32) {
      stage_b_tile_async(sB[cur ^ 1], bbase + (kk + 1) * 32, D_, tid);
      WAIT_ASYNCCNT_2();
    } else {
      WAIT_ASYNCCNT_0();
    }
    __syncthreads();
    __builtin_prefetch(abase + kk * 32 + 128, 0, 0);
    v16h a = load_frag_a(abase + kk * 32, D_, lane);
#pragma unroll
    for (int j = 0; j < 4; ++j) acc[j] = wmma16(a, load_frag_b_lds(sB[cur], j, lane), acc[j]);
    __syncthreads();
  }

  const int n     = lane & 15;
  const int rbase = (lane & 16) ? 8 : 0;
#pragma unroll
  for (int j = 0; j < 4; ++j) {
    const float bv = bias[h * HD_ + j * 16 + n];
#pragma unroll
    for (int r = 0; r < 8; ++r) acc[j][r] += bv;
  }

  const int bh = b * H_ + h;
  if (ROPE) {
    // out[(bh*T + t)*HD + d], rotate (d, d+32) pairs: tiles j and j+2
#pragma unroll
    for (int j = 0; j < 2; ++j) {
#pragma unroll
      for (int r = 0; r < 8; ++r) {
        const int d = j * 16 + n;
        const int t = tloc0 + rbase + r;
        const float cv = cosp[t * HALF_ + d];
        const float sv = sinp[t * HALF_ + d];
        const float x1 = acc[j][r];
        const float x2 = acc[j + 2][r];
        half_t* dst = outp + ((size_t)bh * T_ + t) * HD_;
        dst[d]         = (half_t)(x1 * cv - x2 * sv);
        dst[d + HALF_] = (half_t)(x2 * cv + x1 * sv);
      }
    }
  } else {
    // transposed store: out[(bh*HD + d)*T + t]
#pragma unroll
    for (int j = 0; j < 4; ++j) {
#pragma unroll
      for (int r = 0; r < 8; ++r) {
        const int d = j * 16 + n;
        const int t = tloc0 + rbase + r;
        outp[((size_t)bh * HD_ + d) * T_ + t] = (half_t)acc[j][r];
      }
    }
  }
}

// ---------------------------------------------------------------------------
// Kernel 3: scores  S[q,k] = (Q[q,:] . K[k,:]) * 1/8 + mask[q,k]  (f32 out)
// One wave -> 16 q-rows x 64 k-cols (waves own distinct K-tiles: no sharing,
// so direct global fragment loads; K-loop is only 2 steps).
// ---------------------------------------------------------------------------
__global__ void scores_kernel(const half_t* __restrict__ Qh,
                              const half_t* __restrict__ Kh,
                              const float* __restrict__ mask,
                              float* __restrict__ attnw) {
  const int lane   = threadIdx.x & 31;
  const int wave   = threadIdx.x >> 5;
  const int qtile  = blockIdx.x;               // 0..T/16-1
  const int kgroup = blockIdx.y * 4 + wave;    // 0..T/64-1
  const int bh     = blockIdx.z;               // 0..B*H-1
  const int qbase  = qtile * 16;
  const int kbase  = kgroup * 64;

  const half_t* qb = Qh + ((size_t)bh * T_ + qbase) * HD_;
  const half_t* kb = Kh + ((size_t)bh * T_ + kbase) * HD_;

  v8f acc[4] = {};
#pragma unroll
  for (int d0 = 0; d0 < HD_; d0 += 32) {
    v16h a = load_frag_a(qb + d0, HD_, lane);
#pragma unroll
    for (int j = 0; j < 4; ++j) {
      v16h bf = load_frag_b(kb + (size_t)(j * 16) * HD_ + d0, HD_, lane);
      acc[j] = wmma16(a, bf, acc[j]);
    }
  }

  const float scale = 0.125f;  // 1/sqrt(64)
  const int n     = lane & 15;
  const int rbase = (lane & 16) ? 8 : 0;
  float* obase = attnw + (size_t)bh * T_ * T_;
#pragma unroll
  for (int j = 0; j < 4; ++j) {
    const int kc = kbase + j * 16 + n;
#pragma unroll
    for (int r = 0; r < 8; ++r) {
      const int q = qbase + rbase + r;
      obase[(size_t)q * T_ + kc] = acc[j][r] * scale + mask[(size_t)q * T_ + kc];
    }
  }
}

// ---------------------------------------------------------------------------
// Kernel 4: in-place row softmax over T=2048 columns. 256 threads/row.
// ---------------------------------------------------------------------------
__global__ void softmax_kernel(float* __restrict__ attnw) {
  float* p = attnw + (size_t)blockIdx.x * T_;
  const int tid = threadIdx.x;
  __shared__ float red[256];

  float vals[8];
  float m = -3.0e38f;
#pragma unroll
  for (int i = 0; i < 8; ++i) {
    vals[i] = p[tid + i * 256];
    m = fmaxf(m, vals[i]);
  }
  red[tid] = m;
  __syncthreads();
  for (int s = 128; s > 0; s >>= 1) {
    if (tid < s) red[tid] = fmaxf(red[tid], red[tid + s]);
    __syncthreads();
  }
  m = red[0];
  __syncthreads();

  float sum = 0.f;
#pragma unroll
  for (int i = 0; i < 8; ++i) {
    vals[i] = __expf(vals[i] - m);
    sum += vals[i];
  }
  red[tid] = sum;
  __syncthreads();
  for (int s = 128; s > 0; s >>= 1) {
    if (tid < s) red[tid] += red[tid + s];
    __syncthreads();
  }
  const float inv = 1.0f / red[0];
#pragma unroll
  for (int i = 0; i < 8; ++i) p[tid + i * 256] = vals[i] * inv;
}

// ---------------------------------------------------------------------------
// Kernel 5: ctx[t, h*64+d] = sum_k P[q=t,k] * V[k,d].  A = P (f32, converted
// in-register). B = V^T rows, shared by all 4 waves of the block -> async
// LDS staging, double buffered. One wave -> 16 t x 64 d, K-loop over T.
// ---------------------------------------------------------------------------
__global__ void ctx_kernel(const float* __restrict__ attnw,
                           const half_t* __restrict__ VhT,
                           half_t* __restrict__ ctxh) {
  __shared__ half_t sB[2][64][BROW_];

  const int tid   = threadIdx.x;
  const int lane  = tid & 31;
  const int wave  = tid >> 5;
  const int ttile = blockIdx.x * 4 + wave;     // 0..T/16-1 (per bh)
  const int bh    = blockIdx.y;
  const int b     = bh / H_;
  const int h     = bh - b * H_;
  const int tloc  = ttile * 16;

  const float*  ab = attnw + ((size_t)bh * T_ + tloc) * T_;
  const half_t* vb = VhT + (size_t)bh * HD_ * T_;

  v8f acc[4] = {};
  stage_b_tile_async(sB[0], vb, T_, tid);
  for (int kk = 0; kk < T_ / 32; ++kk) {
    const int cur = kk & 1;
    if (kk + 1 < T_ / 32) {
      stage_b_tile_async(sB[cur ^ 1], vb + (kk + 1) * 32, T_, tid);
      WAIT_ASYNCCNT_2();
    } else {
      WAIT_ASYNCCNT_0();
    }
    __syncthreads();
    __builtin_prefetch(ab + kk * 32 + 128, 0, 0);
    v16h a = load_frag_a_f32(ab + kk * 32, T_, lane);
#pragma unroll
    for (int j = 0; j < 4; ++j) acc[j] = wmma16(a, load_frag_b_lds(sB[cur], j, lane), acc[j]);
    __syncthreads();
  }

  const int n     = lane & 15;
  const int rbase = (lane & 16) ? 8 : 0;
#pragma unroll
  for (int j = 0; j < 4; ++j) {
#pragma unroll
    for (int r = 0; r < 8; ++r) {
      const int d = j * 16 + n;
      const int t = tloc + rbase + r;
      ctxh[((size_t)(b * T_ + t)) * D_ + h * HD_ + d] = (half_t)acc[j][r];
    }
  }
}

// ---------------------------------------------------------------------------
// Kernel 6: out[t,e] = sum_d ctx[t,d] * Wo[e,d] + bo[e]   (f32 out)
// Wo slice shared by all 4 waves -> async LDS staging, double buffered.
// ---------------------------------------------------------------------------
__global__ void outproj_kernel(const half_t* __restrict__ ctxh,
                               const half_t* __restrict__ Woh,
                               const float* __restrict__ bo,
                               float* __restrict__ outp) {
  __shared__ half_t sB[2][64][BROW_];

  const int tid   = threadIdx.x;
  const int lane  = tid & 31;
  const int wave  = tid >> 5;
  const int ttile = blockIdx.x * 4 + wave;     // 0..BT/16-1
  const int eg    = blockIdx.y;                // 0..D/64-1
  const int tbase = ttile * 16;

  const half_t* abase = ctxh + (size_t)tbase * D_;
  const half_t* bbase = Woh + (size_t)(eg * 64) * D_;

  v8f acc[4] = {};
  stage_b_tile_async(sB[0], bbase, D_, tid);
  for (int kk = 0; kk < D_ / 32; ++kk) {
    const int cur = kk & 1;
    if (kk + 1 < D_ / 32) {
      stage_b_tile_async(sB[cur ^ 1], bbase + (kk + 1) * 32, D_, tid);
      WAIT_ASYNCCNT_2();
    } else {
      WAIT_ASYNCCNT_0();
    }
    __syncthreads();
    __builtin_prefetch(abase + kk * 32 + 128, 0, 0);
    v16h a = load_frag_a(abase + kk * 32, D_, lane);
#pragma unroll
    for (int j = 0; j < 4; ++j) acc[j] = wmma16(a, load_frag_b_lds(sB[cur], j, lane), acc[j]);
    __syncthreads();
  }

  const int n     = lane & 15;
  const int rbase = (lane & 16) ? 8 : 0;
#pragma unroll
  for (int j = 0; j < 4; ++j) {
    const int e = eg * 64 + j * 16 + n;
    const float bv = bo[e];
#pragma unroll
    for (int r = 0; r < 8; ++r) {
      const int t = tbase + rbase + r;
      outp[(size_t)t * D_ + e] = acc[j][r] + bv;
    }
  }
}

// ---------------------------------------------------------------------------
// Launch
// ---------------------------------------------------------------------------
extern "C" void kernel_launch(void* const* d_in, const int* in_sizes, int n_in,
                              void* d_out, int out_size, void* d_ws,
                              size_t ws_size, hipStream_t stream) {
  (void)in_sizes; (void)n_in; (void)out_size; (void)ws_size;

  const float* x    = (const float*)d_in[0];
  const float* mask = (const float*)d_in[1];
  const float* cosp = (const float*)d_in[2];
  const float* sinp = (const float*)d_in[3];
  const float* Wq   = (const float*)d_in[4];
  const float* bq   = (const float*)d_in[5];
  const float* Wk   = (const float*)d_in[6];
  const float* bk   = (const float*)d_in[7];
  const float* Wv   = (const float*)d_in[8];
  const float* bv   = (const float*)d_in[9];
  const float* Wo   = (const float*)d_in[10];
  const float* bo   = (const float*)d_in[11];

  float* outp  = (float*)d_out;
  float* attnw = outp + (size_t)BT_ * D_;   // attn_weights region of d_out

  // Workspace carve-up (all f16): ~48 MB total.
  half_t* ws  = (half_t*)d_ws;
  size_t  off = 0;
  half_t* xh   = ws + off; off += (size_t)BT_ * D_;   // 4M
  half_t* Wqh  = ws + off; off += (size_t)D_ * D_;    // 1M
  half_t* Wkh  = ws + off; off += (size_t)D_ * D_;
  half_t* Wvh  = ws + off; off += (size_t)D_ * D_;
  half_t* Woh  = ws + off; off += (size_t)D_ * D_;
  half_t* Qh   = ws + off; off += (size_t)B_ * H_ * T_ * HD_;  // 4M
  half_t* Kh   = ws + off; off += (size_t)B_ * H_ * T_ * HD_;
  half_t* VhT  = ws + off; off += (size_t)B_ * H_ * HD_ * T_;
  half_t* ctxh = ws + off; off += (size_t)BT_ * D_;

  // 1) f32 -> f16 conversions
  cvt_f16_kernel<<<512, 256, 0, stream>>>(x,  xh,  BT_ * D_);
  cvt_f16_kernel<<<256, 256, 0, stream>>>(Wq, Wqh, D_ * D_);
  cvt_f16_kernel<<<256, 256, 0, stream>>>(Wk, Wkh, D_ * D_);
  cvt_f16_kernel<<<256, 256, 0, stream>>>(Wv, Wvh, D_ * D_);
  cvt_f16_kernel<<<256, 256, 0, stream>>>(Wo, Woh, D_ * D_);

  // 2) projections (+bias, +RoPE for Q/K, transposed store for V)
  dim3 gp(BT_ / 64, H_);
  proj_kernel<1><<<gp, 128, 0, stream>>>(xh, Wqh, bq, cosp, sinp, Qh);
  proj_kernel<1><<<gp, 128, 0, stream>>>(xh, Wkh, bk, cosp, sinp, Kh);
  proj_kernel<0><<<gp, 128, 0, stream>>>(xh, Wvh, bv, cosp, sinp, VhT);

  // 3) scores = QK^T * scale + mask  (writes attn region of d_out)
  scores_kernel<<<dim3(T_ / 16, T_ / 256, B_ * H_), 128, 0, stream>>>(
      Qh, Kh, mask, attnw);

  // 4) softmax rows in place
  softmax_kernel<<<B_ * H_ * T_, 256, 0, stream>>>(attnw);

  // 5) ctx = P @ V
  ctx_kernel<<<dim3(T_ / 64, B_ * H_), 128, 0, stream>>>(attnw, VhT, ctxh);

  // 6) out = ctx @ Wo^T + bo
  outproj_kernel<<<dim3(BT_ / 64, D_ / HD_), 128, 0, stream>>>(
      ctxh, Woh, bo, outp);
}
